// Attn_Enc_Dec_26448408609465
// MI455X (gfx1250) — compile-verified, matrix operand order checked by reading, not run on hardware
//
#include <hip/hip_runtime.h>
#include <hip/hip_bf16.h>

typedef _Float16 half_t;
typedef __attribute__((ext_vector_type(16))) _Float16 v16h;
typedef __attribute__((ext_vector_type(8)))  _Float16 v8h;
typedef __attribute__((ext_vector_type(8)))  float    v8f;

#define BATCH 64
#define HDIM  256
#define EDIM  128
#define GATES 1024
#define TSTEPS 64
#define NLAYER 4
#define NGROUP 11
#define NTGT 3

// ---------------- WMMA helpers ----------------

__device__ inline v8f wmma16(v16h a, v16h b, v8f c) {
    // D = A(16x32 f16) * B(32x16 f16) + C(16x16 f32)
    return __builtin_amdgcn_wmma_f32_16x16x32_f16(
        false, a, false, b, (short)0, c, false, false);
}

__device__ inline float sigf(float x) { return 1.0f / (1.0f + __expf(-x)); }
__device__ inline float tanh_fast(float x) { return 2.0f * sigf(2.0f * x) - 1.0f; }

// Accumulate X(64 x LDK) * W^T slice into acc[quadrant][ntile] for M-tile mt.
// Wave wv owns gate columns {q*256 + wv*32 .. +32} for q = 0..3.
// KT/LDK are compile-time so every load offset folds to an immediate; the
// per-fragment packed-weight base is readfirstlane'd into SGPRs so B loads are
// saddr + lane*32B + imm (no per-kt 64-bit VALU address math).
template<int KT, int LDK>
__device__ inline void mma_k(const half_t* __restrict__ X,
                             const half_t* __restrict__ Wp,
                             int mt, int wv, int lane, v8f (&acc)[4][2]) {
    const half_t* rowp = X + (mt * 16 + (lane & 15)) * LDK + ((lane >> 4) << 3);
    int fb[4][2];
#pragma unroll
    for (int q = 0; q < 4; ++q)
#pragma unroll
        for (int j = 0; j < 2; ++j)
            fb[q][j] = __builtin_amdgcn_readfirstlane(
                ((q * 16 + wv * 2 + j) * KT) << 9);   // halves
    const half_t* wl = Wp + lane * 16;
#pragma unroll
    for (int kt = 0; kt < KT; ++kt) {
        union { v16h v; v8h h[2]; } ua;
        ua.h[0] = *(const v8h*)(rowp + kt * 32);
        ua.h[1] = *(const v8h*)(rowp + kt * 32 + 16);
#pragma unroll
        for (int q = 0; q < 4; ++q)
#pragma unroll
            for (int j = 0; j < 2; ++j) {
                v16h b = *(const v16h*)(wl + fb[q][j] + (kt << 9));
                acc[q][j] = wmma16(ua.v, b, acc[q][j]);
            }
    }
}

// ---------------- pack / prep kernels ----------------

// Pack W (row-major [1024][K] f32) into WMMA-B fragment layout, f16.
__global__ __launch_bounds__(256) void pack_w(const float* __restrict__ W,
                                              half_t* __restrict__ out, int K) {
    int KT = K / 32;
    int total = 64 * KT * 512;
    for (int tid = blockIdx.x * blockDim.x + threadIdx.x; tid < total;
         tid += gridDim.x * blockDim.x) {
        int j    = tid & 15;
        int l    = (tid >> 4) & 31;
        int frag = tid >> 9;
        int kt   = frag % KT;
        int nt   = frag / KT;
        int n = nt * 16 + (l & 15);
        int k = kt * 32 + ((j >> 3) << 4) + ((l >> 4) << 3) + (j & 7);
        out[tid] = (half_t)W[(size_t)n * K + k];
    }
}

__global__ __launch_bounds__(256) void add_bias(const float* __restrict__ a,
                                                const float* __restrict__ b,
                                                float* __restrict__ out, int n) {
    for (int i = blockIdx.x * blockDim.x + threadIdx.x; i < n;
         i += gridDim.x * blockDim.x)
        out[i] = a[i] + b[i];
}

// emb[g][t][b][e] = relu(x[b][t][g][:] . W[e][:] + bias[e]), f16
__global__ __launch_bounds__(256) void embed_enc(const float* __restrict__ x,
                                                 const float* __restrict__ W,
                                                 const float* __restrict__ bias,
                                                 half_t* __restrict__ emb) {
    int total = NGROUP * TSTEPS * BATCH * EDIM;
    for (int idx = blockIdx.x * blockDim.x + threadIdx.x; idx < total;
         idx += gridDim.x * blockDim.x) {
        int e = idx & 127;
        int b = (idx >> 7) & 63;
        int t = (idx >> 13) & 63;
        int g = idx >> 19;
        const float* xp = x + (((size_t)b * TSTEPS + t) * NGROUP + g) * 4;
        const float* w  = W + e * 4;
        float s = bias[e] + xp[0]*w[0] + xp[1]*w[1] + xp[2]*w[2] + xp[3]*w[3];
        emb[idx] = (half_t)fmaxf(s, 0.0f);
    }
}

// ---------------- encoder LSTM (one workgroup per group) ----------------

__global__ __launch_bounds__(256) void enc_lstm(
    const half_t* __restrict__ emb,    // [G][T][B][E]
    const half_t* __restrict__ WihP,   // packed (layer offsets below)
    const half_t* __restrict__ WhhP,   // packed, 262144 halves per layer
    const float*  __restrict__ bias,   // [4][1024]
    half_t* __restrict__ enc_outs,     // [G][T][B][H]
    half_t* __restrict__ encH,         // [G][L][B][H]
    half_t* __restrict__ encC)         // [G][L][B][H]
{
    extern __shared__ char smem[];
    half_t* hbuf = (half_t*)smem;                       // [4][64][256]
    half_t* cbuf = hbuf + NLAYER * BATCH * HDIM;        // [4][64][256]
    half_t* xemb = cbuf + NLAYER * BATCH * HDIM;        // [64][128]

    const int g    = blockIdx.x;
    const int tid  = threadIdx.x;
    const int wv   = tid >> 5;
    const int lane = tid & 31;
    const int nslice = wv * 32;

    for (int i = tid; i < NLAYER * BATCH * HDIM; i += 256) {
        hbuf[i] = (half_t)0.0f;
        cbuf[i] = (half_t)0.0f;
    }
    __syncthreads();

    for (int t = 0; t < TSTEPS; ++t) {
        // stage this step's embedding into LDS
        const half_t* src = emb + (((size_t)g * TSTEPS + t) * BATCH) * EDIM;
        for (int i = tid; i < BATCH * EDIM / 8; i += 256)
            ((v8h*)xemb)[i] = ((const v8h*)src)[i];
        // prefetch next step's embedding tile (global_prefetch_b8 path)
        if (t + 1 < TSTEPS)
            __builtin_prefetch(src + BATCH * EDIM + tid * 64, 0, 1);
        __syncthreads();

        for (int l = 0; l < NLAYER; ++l) {
            const half_t* Hl = hbuf + l * BATCH * HDIM;
            const half_t* Whh = WhhP + (size_t)l * 262144;
            const float* bL   = bias + l * GATES;
            if (l < NLAYER - 1)
                __builtin_prefetch(WhhP + (size_t)(l + 1) * 262144 + tid * 64, 0, 1);

            v8f hreg[4][2];
            for (int mt = 0; mt < 4; ++mt) {
                v8f acc[4][2];
#pragma unroll
                for (int q = 0; q < 4; ++q)
#pragma unroll
                    for (int j = 0; j < 2; ++j) acc[q][j] = {};

                if (l == 0) {
                    mma_k<4, 128>(xemb, WihP, mt, wv, lane, acc);
                } else {
                    mma_k<8, 256>(hbuf + (l - 1) * BATCH * HDIM,
                                  WihP + 131072 + (size_t)(l - 1) * 262144,
                                  mt, wv, lane, acc);
                }
                mma_k<8, 256>(Hl, Whh, mt, wv, lane, acc);   // h @ Whh^T

                const int m0 = mt * 16 + ((lane >> 4) << 3);
#pragma unroll
                for (int j = 0; j < 2; ++j) {
                    const int n  = nslice + j * 16 + (lane & 15);
                    const float bI = bL[n];
                    const float bF = bL[256 + n];
                    const float bG = bL[512 + n];
                    const float bO = bL[768 + n];
#pragma unroll
                    for (int r = 0; r < 8; ++r) {
                        float iv = acc[0][j][r] + bI;
                        float fv = acc[1][j][r] + bF;
                        float gv = acc[2][j][r] + bG;
                        float ov = acc[3][j][r] + bO;
                        int cidx = l * BATCH * HDIM + (m0 + r) * HDIM + n;
                        float c2 = sigf(fv) * (float)cbuf[cidx]
                                 + sigf(iv) * tanh_fast(gv);
                        cbuf[cidx] = (half_t)c2;
                        hreg[mt][j][r] = sigf(ov) * tanh_fast(c2);
                    }
                }
            }
            __syncthreads();   // all reads of hbuf[l], hbuf[l-1] done
#pragma unroll
            for (int mt = 0; mt < 4; ++mt)
#pragma unroll
                for (int j = 0; j < 2; ++j)
#pragma unroll
                    for (int r = 0; r < 8; ++r) {
                        int m = mt * 16 + ((lane >> 4) << 3) + r;
                        int n = nslice + j * 16 + (lane & 15);
                        hbuf[l * BATCH * HDIM + m * HDIM + n] =
                            (half_t)hreg[mt][j][r];
                    }
            __syncthreads();   // new h visible
        }

        // top-layer h -> enc_outs
        const half_t* h3 = hbuf + 3 * BATCH * HDIM;
        half_t* dst = enc_outs + (((size_t)g * TSTEPS + t) * BATCH) * HDIM;
        for (int i = tid; i < BATCH * HDIM / 8; i += 256)
            ((v8h*)dst)[i] = ((const v8h*)h3)[i];
        __syncthreads();
    }

    // final h, c for all layers
    half_t* dh = encH + (size_t)g * NLAYER * BATCH * HDIM;
    half_t* dc = encC + (size_t)g * NLAYER * BATCH * HDIM;
    for (int i = tid; i < NLAYER * BATCH * HDIM / 8; i += 256) {
        ((v8h*)dh)[i] = ((const v8h*)hbuf)[i];
        ((v8h*)dc)[i] = ((const v8h*)cbuf)[i];
    }
}

// ---------------- attention (one workgroup per batch element) ----------------

__global__ __launch_bounds__(256) void attention_k(
    const half_t* __restrict__ enc_outs,  // [G][T][B][H]
    const half_t* __restrict__ encH,      // [G][L][B][H]
    const float*  __restrict__ attn_W,    // [512]: [0:256]=wa_h, [256:512]=wa_e
    const float*  __restrict__ attn_b,
    const int*    __restrict__ to_ptr,
    float* __restrict__ attn_out)         // [B][H]
{
    __shared__ float sw[NGROUP * TSTEPS];  // 704 scores
    __shared__ float red[256];
    const int b = blockIdx.x, tid = threadIdx.x;
    const int to = to_ptr[0];

    const half_t* hq = encH + (((size_t)to * NLAYER + 0) * BATCH + b) * HDIM;
    float shq = attn_b[0];
    for (int h = 0; h < HDIM; ++h) shq += (float)hq[h] * attn_W[h];

    for (int p = tid; p < NGROUP * TSTEPS; p += 256) {
        int gg = p / TSTEPS, tt = p % TSTEPS;
        const half_t* eo = enc_outs + (((size_t)gg * TSTEPS + tt) * BATCH + b) * HDIM;
        float s = 0.0f;
        for (int h = 0; h < HDIM; ++h) s += (float)eo[h] * attn_W[256 + h];
        sw[p] = s + shq;
    }
    __syncthreads();

    float mx = -1e30f;
    for (int p = tid; p < NGROUP * TSTEPS; p += 256) mx = fmaxf(mx, sw[p]);
    red[tid] = mx; __syncthreads();
    for (int s = 128; s > 0; s >>= 1) {
        if (tid < s) red[tid] = fmaxf(red[tid], red[tid + s]);
        __syncthreads();
    }
    mx = red[0]; __syncthreads();

    float sm = 0.0f;
    for (int p = tid; p < NGROUP * TSTEPS; p += 256) {
        float e = __expf(sw[p] - mx);
        sw[p] = e; sm += e;
    }
    red[tid] = sm; __syncthreads();
    for (int s = 128; s > 0; s >>= 1) {
        if (tid < s) red[tid] += red[tid + s];
        __syncthreads();
    }
    float inv = 1.0f / red[0];
    __syncthreads();

    if (tid < HDIM) {
        int h = tid;
        float a = 0.0f;
        for (int p = 0; p < NGROUP * TSTEPS; ++p) {
            int gg = p / TSTEPS, tt = p % TSTEPS;
            a += sw[p] *
                 (float)enc_outs[(((size_t)gg * TSTEPS + tt) * BATCH + b) * HDIM + h];
        }
        attn_out[b * HDIM + h] = a * inv;
    }
}

// ---------------- decoder (one workgroup per target) ----------------

__global__ __launch_bounds__(256) void dec_lstm(
    const float*  __restrict__ x,        // [B][T][G][F]
    const float*  __restrict__ attn_sum, // [B][H] f32
    const half_t* __restrict__ encH,     // [G][L][B][H]
    const half_t* __restrict__ encC,     // [G][L][B][H]
    const half_t* __restrict__ WihP,     // packed dec Wih (l0 K=384)
    const half_t* __restrict__ WhhP,     // packed dec Whh
    const float*  __restrict__ bias,     // [4][1024]
    const float*  __restrict__ emb_W,    // [128][4]
    const float*  __restrict__ emb_b,    // [128]
    const float*  __restrict__ out_W,    // [4][256]
    const float*  __restrict__ out_b,    // [4]
    const int*    __restrict__ to_ptr,
    float* __restrict__ out)             // [B][TT][NT][F]
{
    extern __shared__ char smem[];
    half_t* xbuf  = (half_t*)smem;                 // [64][384]: [0:256)=attn, [256:384)=e
    half_t* hnext = xbuf + BATCH * 384;            // [64][256]
    float*  curin = (float*)(hnext + BATCH * HDIM);// [64][4]

    const int j    = blockIdx.x;        // target index
    const int tid  = threadIdx.x;
    const int wv   = tid >> 5;
    const int lane = tid & 31;
    const int nslice = wv * 32;
    const int to = to_ptr[0];

    // constant attn part of layer-0 input, f32 -> f16
    for (int i = tid; i < BATCH * HDIM; i += 256) {
        int b = i >> 8, h = i & 255;
        xbuf[b * 384 + h] = (half_t)attn_sum[i];
    }
    // initial decoder input = x[:, T-1, to+j, :]
    for (int i = tid; i < BATCH * 4; i += 256) {
        int b = i >> 2, f = i & 3;
        curin[i] = x[(((size_t)b * TSTEPS + (TSTEPS - 1)) * NGROUP + (to + j)) * 4 + f];
    }
    __syncthreads();

    for (int t = 0; t < TSTEPS; ++t) {
        // e = relu(curin @ emb_W^T + emb_b) -> xbuf cols [256:384)
        for (int i = tid; i < BATCH * EDIM; i += 256) {
            int b = i >> 7, e = i & 127;
            const float* w  = emb_W + e * 4;
            const float* ci = curin + b * 4;
            float s = emb_b[e] + ci[0]*w[0] + ci[1]*w[1] + ci[2]*w[2] + ci[3]*w[3];
            xbuf[b * 384 + 256 + e] = (half_t)fmaxf(s, 0.0f);
        }
        __syncthreads();

        for (int l = 0; l < NLAYER; ++l) {
            const half_t* Whh = WhhP + (size_t)l * 262144;
            const half_t* Hc  = encH + ((size_t)to * NLAYER + l) * BATCH * HDIM;
            const half_t* Cc  = encC + ((size_t)to * NLAYER + l) * BATCH * HDIM;
            const float*  bL  = bias + l * GATES;

            v8f hreg[4][2];
            for (int mt = 0; mt < 4; ++mt) {
                v8f acc[4][2];
#pragma unroll
                for (int q = 0; q < 4; ++q)
#pragma unroll
                    for (int jj = 0; jj < 2; ++jj) acc[q][jj] = {};

                if (l == 0) {
                    mma_k<12, 384>(xbuf, WihP, mt, wv, lane, acc);
                } else {
                    mma_k<8, 256>(hnext, WihP + 393216 + (size_t)(l - 1) * 262144,
                                  mt, wv, lane, acc);
                }
                mma_k<8, 256>(Hc, Whh, mt, wv, lane, acc);   // h_const @ Whh^T

                const int m0 = mt * 16 + ((lane >> 4) << 3);
#pragma unroll
                for (int jj = 0; jj < 2; ++jj) {
                    const int n  = nslice + jj * 16 + (lane & 15);
                    const float bI = bL[n];
                    const float bF = bL[256 + n];
                    const float bG = bL[512 + n];
                    const float bO = bL[768 + n];
#pragma unroll
                    for (int r = 0; r < 8; ++r) {
                        float iv = acc[0][jj][r] + bI;
                        float fv = acc[1][jj][r] + bF;
                        float gv = acc[2][jj][r] + bG;
                        float ov = acc[3][jj][r] + bO;
                        float c0 = (float)Cc[(m0 + r) * HDIM + n];
                        float c2 = sigf(fv) * c0 + sigf(iv) * tanh_fast(gv);
                        hreg[mt][jj][r] = sigf(ov) * tanh_fast(c2);
                    }
                }
            }
            __syncthreads();
#pragma unroll
            for (int mt = 0; mt < 4; ++mt)
#pragma unroll
                for (int jj = 0; jj < 2; ++jj)
#pragma unroll
                    for (int r = 0; r < 8; ++r) {
                        int m = mt * 16 + ((lane >> 4) << 3) + r;
                        int n = nslice + jj * 16 + (lane & 15);
                        hnext[m * HDIM + n] = (half_t)hreg[mt][jj][r];
                    }
            __syncthreads();
        }

        // out = hnext @ out_W^T + out_b ; feed back + emit
        for (int i = tid; i < BATCH * 4; i += 256) {
            int b = i >> 2, f = i & 3;
            const float*  w  = out_W + f * HDIM;
            const half_t* hp = hnext + b * HDIM;
            float s = out_b[f];
            for (int h = 0; h < HDIM; ++h) s += (float)hp[h] * w[h];
            curin[i] = s;
            out[(((size_t)b * TSTEPS + t) * NTGT + j) * 4 + f] = s;
        }
        __syncthreads();
    }
}

// ---------------- host side ----------------

extern "C" void kernel_launch(void* const* d_in, const int* in_sizes, int n_in,
                              void* d_out, int out_size, void* d_ws, size_t ws_size,
                              hipStream_t stream) {
    const float* x         = (const float*)d_in[0];
    const float* enc_lin_W = (const float*)d_in[2];
    const float* enc_lin_b = (const float*)d_in[3];
    const float* enc_Wih0  = (const float*)d_in[4];
    const float* enc_Wihs  = (const float*)d_in[5];
    const float* enc_Whh   = (const float*)d_in[6];
    const float* enc_bih   = (const float*)d_in[7];
    const float* enc_bhh   = (const float*)d_in[8];
    const float* dec_emb_W = (const float*)d_in[9];
    const float* dec_emb_b = (const float*)d_in[10];
    const float* attn_W    = (const float*)d_in[11];
    const float* attn_b    = (const float*)d_in[12];
    const float* dec_Wih0  = (const float*)d_in[13];
    const float* dec_Wihs  = (const float*)d_in[14];
    const float* dec_Whh   = (const float*)d_in[15];
    const float* dec_bih   = (const float*)d_in[16];
    const float* dec_bhh   = (const float*)d_in[17];
    const float* out_W     = (const float*)d_in[18];
    const float* out_b     = (const float*)d_in[19];
    const int*   to_ptr    = (const int*)d_in[20];
    float* out = (float*)d_out;

    // workspace carve-up (halves / floats)
    char* ws = (char*)d_ws;
    half_t* WihEp = (half_t*)ws;           ws += (131072 + 3 * 262144) * 2;
    half_t* WhhEp = (half_t*)ws;           ws += (4 * 262144) * 2;
    half_t* WihDp = (half_t*)ws;           ws += (393216 + 3 * 262144) * 2;
    half_t* WhhDp = (half_t*)ws;           ws += (4 * 262144) * 2;
    float*  biasE = (float*)ws;            ws += 4096 * 4;
    float*  biasD = (float*)ws;            ws += 4096 * 4;
    half_t* emb   = (half_t*)ws;           ws += (size_t)NGROUP*TSTEPS*BATCH*EDIM * 2;
    half_t* eouts = (half_t*)ws;           ws += (size_t)NGROUP*TSTEPS*BATCH*HDIM * 2;
    half_t* encH  = (half_t*)ws;           ws += (size_t)NGROUP*NLAYER*BATCH*HDIM * 2;
    half_t* encC  = (half_t*)ws;           ws += (size_t)NGROUP*NLAYER*BATCH*HDIM * 2;
    float*  attnS = (float*)ws;            ws += BATCH * HDIM * 4;

    // 1) pack weights to f16 WMMA-B layout
    pack_w<<<512, 256, 0, stream>>>(enc_Wih0, WihEp, 128);
    for (int l = 0; l < 3; ++l)
        pack_w<<<1024, 256, 0, stream>>>(enc_Wihs + (size_t)l * 1024 * 256,
                                         WihEp + 131072 + (size_t)l * 262144, 256);
    for (int l = 0; l < 4; ++l)
        pack_w<<<1024, 256, 0, stream>>>(enc_Whh + (size_t)l * 1024 * 256,
                                         WhhEp + (size_t)l * 262144, 256);
    pack_w<<<1536, 256, 0, stream>>>(dec_Wih0, WihDp, 384);
    for (int l = 0; l < 3; ++l)
        pack_w<<<1024, 256, 0, stream>>>(dec_Wihs + (size_t)l * 1024 * 256,
                                         WihDp + 393216 + (size_t)l * 262144, 256);
    for (int l = 0; l < 4; ++l)
        pack_w<<<1024, 256, 0, stream>>>(dec_Whh + (size_t)l * 1024 * 256,
                                         WhhDp + (size_t)l * 262144, 256);
    add_bias<<<16, 256, 0, stream>>>(enc_bih, enc_bhh, biasE, 4096);
    add_bias<<<16, 256, 0, stream>>>(dec_bih, dec_bhh, biasD, 4096);

    // 2) encoder embedding
    embed_enc<<<(NGROUP*TSTEPS*BATCH*EDIM + 255) / 256, 256, 0, stream>>>(
        x, enc_lin_W, enc_lin_b, emb);

    // 3) encoder LSTM: one WGP per group; h/c resident in LDS (272 KB)
    size_t enc_smem = (size_t)(NLAYER*BATCH*HDIM*2 + BATCH*EDIM) * sizeof(half_t);
    enc_lstm<<<NGROUP, 256, enc_smem, stream>>>(emb, WihEp, WhhEp, biasE,
                                                eouts, encH, encC);

    // 4) attention
    attention_k<<<BATCH, 256, 0, stream>>>(eouts, encH, attn_W, attn_b,
                                           to_ptr, attnS);

    // 5) decoder: one WGP per target
    size_t dec_smem = (size_t)(BATCH*384 + BATCH*HDIM) * sizeof(half_t)
                    + (size_t)BATCH * 4 * sizeof(float);
    dec_lstm<<<NTGT, 256, dec_smem, stream>>>(x, attnS, encH, encC,
                                              WihDp, WhhDp, biasD,
                                              dec_emb_W, dec_emb_b,
                                              out_W, out_b, to_ptr, out);
}
